// xl_Encoder_52939766890896
// MI455X (gfx1250) — matrix operand order repaired and checked by reading, not compiled, not run on hardware
//
#include <hip/hip_runtime.h>
#include <hip/hip_bf16.h>

// ---------------- constants ----------------
#define DD    512
#define KK    512
#define SEGL  512
#define NLAY  6
#define NSEGS 8
#define FFD   2048
#define MEMR  1024          // REUSE*SEG
#define FULLR 1536          // MEM + SEG
#define NTOK  4096          // NSEG*SEG

typedef __attribute__((ext_vector_type(16))) __bf16      v16bf;
typedef __attribute__((ext_vector_type(8)))  float       v8f;
typedef __attribute__((ext_vector_type(4)))  unsigned int u32x4;
typedef __attribute__((ext_vector_type(8)))  unsigned int u32x8;
typedef int vsi4 __attribute__((vector_size(16)));   // matches builtin's int-vector pointee

#if defined(__gfx1250__) && __has_builtin(__builtin_amdgcn_global_load_async_to_lds_b128)
#define HAVE_ASYNC_LDS 1
#else
#define HAVE_ASYNC_LDS 0
#endif

// ---------------- helpers ----------------
__device__ __forceinline__ unsigned short f2bf(float f) {
    unsigned int u = __builtin_bit_cast(unsigned int, f);
    unsigned int r = u + 0x7fffu + ((u >> 16) & 1u);
    return (unsigned short)(r >> 16);
}

__device__ __forceinline__ v16bf pack16(u32x4 lo, u32x4 hi) {
    u32x8 t;
    t[0] = lo[0]; t[1] = lo[1]; t[2] = lo[2]; t[3] = lo[3];
    t[4] = hi[0]; t[5] = hi[1]; t[6] = hi[2]; t[7] = hi[3];
    return __builtin_bit_cast(v16bf, t);
}

#if HAVE_ASYNC_LDS
__device__ __forceinline__ void async_ld_b128(const unsigned short* g, unsigned short* l) {
    __attribute__((address_space(1))) vsi4* gp =
        (__attribute__((address_space(1))) vsi4*)(unsigned long long)(const void*)g;
    __attribute__((address_space(3))) vsi4* lp =
        (__attribute__((address_space(3))) vsi4*)(unsigned int)(unsigned long long)(void*)l;
    __builtin_amdgcn_global_load_async_to_lds_b128(gp, lp, 0, 0);
}
#endif

__device__ __forceinline__ void async_wait() {
#if HAVE_ASYNC_LDS
#if __has_builtin(__builtin_amdgcn_s_wait_asynccnt)
    __builtin_amdgcn_s_wait_asynccnt(0);
#else
    asm volatile("s_wait_asynccnt 0" ::: "memory");
#endif
#endif
}

// ---------------- WMMA GEMM ----------------
// C[M,N] = A[M,Kd] * (BT ? B[N,Kd]^T : B[Kd,N]), bf16 in, fp32 accum.
// 128 threads = 4 waves; 64x64 tile; K-step 32; double-buffered LDS with
// async global->LDS DMA (ASYNCcnt) for row-major tiles.
// OUTF/OUTBF select fp32/bf16 output at compile time (branch-free epilogue).
// LDS pitch 40 elems (80B = 5x16B) -> 16B-aligned, conflict-free b128 loads.
template<int BT, int RELU, int ACCUM, int OUTF, int OUTBF>
__global__ __launch_bounds__(128) void gemm_wmma(
    const unsigned short* __restrict__ A,
    const unsigned short* __restrict__ B,
    float* __restrict__ C,
    unsigned short* __restrict__ Cbf,
    int M, int N, int Kd)
{
    __shared__ __align__(16) unsigned short lds_a[2][64 * 40];
    __shared__ __align__(16) unsigned short lds_b[2][64 * 40];

    const int tid  = threadIdx.x;
    const int lane = tid & 31;
    const int wave = tid >> 5;
    const int row0 = blockIdx.y * 64;
    const int col0 = blockIdx.x * 64;
    const int nk   = Kd >> 5;

    v8f acc[4] = {};
    unsigned short breg[16];   // non-BT transpose staging registers

    // stage a 64x32 bf16 row-major tile (rows r0.., cols kk..) into pitched LDS
    auto stage_rowmajor = [&](const unsigned short* src, int r0, int kk,
                              unsigned short* dst) {
#if HAVE_ASYNC_LDS
        #pragma unroll
        for (int it = 0; it < 2; ++it) {
            int idx = it * 128 + tid;          // 0..255 chunks of 16B
            int r = idx >> 2, c = idx & 3;
            async_ld_b128(src + (size_t)(r0 + r) * Kd + kk + c * 8,
                          dst + r * 40 + c * 8);
        }
#else
        u32x4 tmp[2];
        #pragma unroll
        for (int it = 0; it < 2; ++it) {
            int idx = it * 128 + tid;
            int r = idx >> 2, c = idx & 3;
            tmp[it] = *(const u32x4*)(src + (size_t)(r0 + r) * Kd + kk + c * 8);
        }
        #pragma unroll
        for (int it = 0; it < 2; ++it) {
            int idx = it * 128 + tid;
            int r = idx >> 2, c = idx & 3;
            *(u32x4*)(dst + r * 40 + c * 8) = tmp[it];
        }
#endif
    };

    // non-BT: B is Kd x N row-major; transpose-stage -> lds[n][k]
    auto loadB_regs = [&](int kk) {
        #pragma unroll
        for (int it = 0; it < 16; ++it) {
            int idx = it * 128 + tid;
            int k = idx >> 6, n = idx & 63;
            breg[it] = B[(size_t)(kk + k) * N + col0 + n];
        }
    };
    auto storeB_regs = [&](unsigned short* dst) {
        #pragma unroll
        for (int it = 0; it < 16; ++it) {
            int idx = it * 128 + tid;
            int k = idx >> 6, n = idx & 63;
            dst[n * 40 + k] = breg[it];
        }
    };

    auto compute = [&](const unsigned short* la, const unsigned short* lb) {
        const int m   = wave * 16 + (lane & 15);
        const int h8  = (lane >> 4) << 3;
        const int k16 = (lane >> 4) << 4;
        u32x4 alo = *(const u32x4*)(la + m * 40 + h8);
        u32x4 ahi = *(const u32x4*)(la + m * 40 + 16 + h8);
        v16bf afrag = pack16(alo, ahi);
        // preload all B fragments...
        v16bf b0, b1, b2, b3;
        {
            const int n0 = (lane & 15);
            b0 = pack16(*(const u32x4*)(lb + (n0 +  0) * 40 + h8),
                        *(const u32x4*)(lb + (n0 +  0) * 40 + 16 + h8));
            b1 = pack16(*(const u32x4*)(lb + (n0 + 16) * 40 + h8),
                        *(const u32x4*)(lb + (n0 + 16) * 40 + 16 + h8));
            b2 = pack16(*(const u32x4*)(lb + (n0 + 32) * 40 + h8),
                        *(const u32x4*)(lb + (n0 + 32) * 40 + 16 + h8));
            b3 = pack16(*(const u32x4*)(lb + (n0 + 48) * 40 + h8),
                        *(const u32x4*)(lb + (n0 + 48) * 40 + 16 + h8));
        }
        // ...force them live simultaneously so the DS waits collapse to one
        // and the four WMMAs issue back-to-back
        asm volatile("" :: "v"(afrag), "v"(b0), "v"(b1), "v"(b2), "v"(b3));
        acc[0] = __builtin_amdgcn_wmma_f32_16x16x32_bf16(
            false, afrag, false, b0, (short)0, acc[0], false, false);
        acc[1] = __builtin_amdgcn_wmma_f32_16x16x32_bf16(
            false, afrag, false, b1, (short)0, acc[1], false, false);
        acc[2] = __builtin_amdgcn_wmma_f32_16x16x32_bf16(
            false, afrag, false, b2, (short)0, acc[2], false, false);
        acc[3] = __builtin_amdgcn_wmma_f32_16x16x32_bf16(
            false, afrag, false, b3, (short)0, acc[3], false, false);
    };

    // ---- prologue: tile 0 -> buffer 0 ----
    stage_rowmajor(A, row0, 0, lds_a[0]);
    if (BT) {
        stage_rowmajor(B, col0, 0, lds_b[0]);
    } else {
        loadB_regs(0);
        storeB_regs(lds_b[0]);
    }
    async_wait();
    __syncthreads();

    // ---- main loop: prefetch t+1 while computing t ----
    for (int t = 0; t < nk; ++t) {
        const int p = t & 1;
        if (t + 1 < nk) {
            const int kk2 = (t + 1) << 5;
            stage_rowmajor(A, row0, kk2, lds_a[1 - p]);
            if (BT) {
                stage_rowmajor(B, col0, kk2, lds_b[1 - p]);
            } else {
                loadB_regs(kk2);
                storeB_regs(lds_b[1 - p]);
            }
        }
        compute(lds_a[p], lds_b[p]);
        if (t + 1 < nk) async_wait();
        __syncthreads();
    }

    // ---- epilogue: C/D layout -> VGPR r holds M = r + 8*(lane>=16), N = lane&15
    const int mbase = row0 + wave * 16 + ((lane >> 4) << 3);
    const int nsub  = lane & 15;
    #pragma unroll
    for (int c = 0; c < 4; ++c) {
        const int col = col0 + c * 16 + nsub;
        #pragma unroll
        for (int r = 0; r < 8; ++r) {
            size_t o = (size_t)(mbase + r) * N + col;
            float v = acc[c][r];
            if (ACCUM) v += C[o];
            if (RELU)  v = fmaxf(v, 0.0f);
            if (OUTF)  C[o] = v;
            if (OUTBF) Cbf[o] = f2bf(v);
        }
    }
}

// ---------------- elementwise / reduction kernels ----------------
__global__ __launch_bounds__(256) void f2bf_kernel(const float* __restrict__ s,
                                                   unsigned short* __restrict__ d, int n) {
    int i = blockIdx.x * 256 + threadIdx.x;
    if (i < n) d[i] = f2bf(s[i]);
}

__global__ __launch_bounds__(256) void compute_R(unsigned short* __restrict__ Rbf) {
    // R = flip(sinusoid_pe(1536, 512), axis=0), stored bf16
    int p = blockIdx.x;                 // 0..1535
    int pos = (FULLR - 1) - p;          // flipped position
    const float nlog = -9.210340371976184f / 512.0f; // -ln(10000)/dim
    #pragma unroll
    for (int t = 0; t < 2; ++t) {
        int c = t * 256 + threadIdx.x;
        float ipair = (float)(c & ~1);
        float ang = (float)pos * __expf(nlog * ipair);
        float v = (c & 1) ? __cosf(ang) : __sinf(ang);
        Rbf[(size_t)p * DD + c] = f2bf(v);
    }
}

__global__ __launch_bounds__(256) void embed_gather(
    const int* __restrict__ src, const float* __restrict__ emb,
    float* __restrict__ state_f, unsigned short* __restrict__ state_bf,
    int seg, int rev)
{
    int p = blockIdx.x;                        // row in segment
    int pos = seg * SEGL + p;
    int tidx = rev ? (NTOK - 1 - pos) : pos;
    int tok = src[tidx];
    #pragma unroll
    for (int t = 0; t < 2; ++t) {
        int c = t * 256 + threadIdx.x;
        float v = emb[(size_t)tok * DD + c];
        state_f[(size_t)p * DD + c] = v;
        state_bf[(size_t)p * DD + c] = f2bf(v);
    }
}

__global__ __launch_bounds__(256) void add_row_bf(
    const float* __restrict__ q, const float* __restrict__ rowv,
    unsigned short* __restrict__ out)
{
    int i = blockIdx.x * 256 + threadIdx.x;    // n = 512*512
    out[i] = f2bf(q[i] + rowv[i & (KK - 1)]);
}

__global__ __launch_bounds__(256) void softmax_rows(
    const float* __restrict__ S, unsigned short* __restrict__ P, int n_valid)
{
    __shared__ float red[256];
    const int row = blockIdx.x;
    const int t = threadIdx.x;
    const float scale = 0.044194173824159216f;  // 1/sqrt(512)
    float vals[6];
    float mx = -1e30f;
    #pragma unroll
    for (int k = 0; k < 6; ++k) {
        int c = k * 256 + t;
        float s = S[(size_t)row * FULLR + c] * scale;
        if (c < MEMR - n_valid) s += -1e9f;
        vals[k] = s;
        mx = fmaxf(mx, s);
    }
    red[t] = mx; __syncthreads();
    for (int o = 128; o > 0; o >>= 1) { if (t < o) red[t] = fmaxf(red[t], red[t + o]); __syncthreads(); }
    mx = red[0]; __syncthreads();
    float sum = 0.0f;
    #pragma unroll
    for (int k = 0; k < 6; ++k) { vals[k] = __expf(vals[k] - mx); sum += vals[k]; }
    red[t] = sum; __syncthreads();
    for (int o = 128; o > 0; o >>= 1) { if (t < o) red[t] += red[t + o]; __syncthreads(); }
    float inv = 1.0f / red[0];
    #pragma unroll
    for (int k = 0; k < 6; ++k) {
        int c = k * 256 + t;
        P[(size_t)row * FULLR + c] = f2bf(vals[k] * inv);
    }
}

__global__ __launch_bounds__(256) void ln_residual(
    const float* __restrict__ X, const float* __restrict__ Y,
    const float* __restrict__ g, const float* __restrict__ b,
    float* __restrict__ out_f, unsigned short* __restrict__ out_bf)
{
    __shared__ float red[256];
    const int row = blockIdx.x;
    const int t = threadIdx.x;
    float x0 = X[(size_t)row * DD + t]       + Y[(size_t)row * DD + t];
    float x1 = X[(size_t)row * DD + 256 + t] + Y[(size_t)row * DD + 256 + t];
    red[t] = x0 + x1; __syncthreads();
    for (int o = 128; o > 0; o >>= 1) { if (t < o) red[t] += red[t + o]; __syncthreads(); }
    float mu = red[0] * (1.0f / 512.0f); __syncthreads();
    float d0 = x0 - mu, d1 = x1 - mu;
    red[t] = d0 * d0 + d1 * d1; __syncthreads();
    for (int o = 128; o > 0; o >>= 1) { if (t < o) red[t] += red[t + o]; __syncthreads(); }
    float rstd = rsqrtf(red[0] * (1.0f / 512.0f) + 1e-5f);
    float o0 = d0 * rstd * g[t]       + b[t];
    float o1 = d1 * rstd * g[256 + t] + b[256 + t];
    out_f[(size_t)row * DD + t] = o0;
    out_f[(size_t)row * DD + 256 + t] = o1;
    if (out_bf) {
        out_bf[(size_t)row * DD + t] = f2bf(o0);
        out_bf[(size_t)row * DD + 256 + t] = f2bf(o1);
    }
}

__global__ __launch_bounds__(256) void write_out(
    const float* __restrict__ state, float* __restrict__ out, int seg, int rev)
{
    int p = blockIdx.x;
    int pos = seg * SEGL + p;
    int orow = rev ? (NTOK - 1 - pos) : pos;
    int ocol = rev ? DD : 0;
    #pragma unroll
    for (int t = 0; t < 2; ++t) {
        int c = t * 256 + threadIdx.x;
        out[(size_t)orow * (2 * DD) + ocol + c] = state[(size_t)p * DD + c];
    }
}

// ---------------- host orchestration ----------------
extern "C" void kernel_launch(void* const* d_in, const int* in_sizes, int n_in,
                              void* d_out, int out_size, void* d_ws, size_t ws_size,
                              hipStream_t stream) {
    (void)in_sizes; (void)n_in; (void)out_size; (void)ws_size;

    const int*   src = (const int*)d_in[0];
    const float* emb = (const float*)d_in[2];
    const float* Wqkvr[2] = { (const float*)d_in[3], (const float*)d_in[9]  };
    const float* Wo[2]    = { (const float*)d_in[4], (const float*)d_in[10] };
    const float* W1[2]    = { (const float*)d_in[5], (const float*)d_in[11] };
    const float* W2[2]    = { (const float*)d_in[6], (const float*)d_in[12] };
    const float* LN[2]    = { (const float*)d_in[7], (const float*)d_in[13] };
    const float* UV[2]    = { (const float*)d_in[8], (const float*)d_in[14] };
    float* out = (float*)d_out;

    // --- workspace layout (deterministic bump allocator) ---
    char* ws = (char*)d_ws;
    size_t off = 0;
    auto alloc = [&](size_t bytes) -> void* {
        off = (off + 255) & ~(size_t)255;
        void* p = ws + off;
        off += bytes;
        return p;
    };
    typedef unsigned short us;
    const size_t nWqkvr = (size_t)NLAY * 4 * DD * KK;
    const size_t nWo    = (size_t)NLAY * KK * DD;
    const size_t nW1    = (size_t)NLAY * DD * FFD;
    const size_t nW2    = (size_t)NLAY * FFD * DD;
    const size_t nPK    = (size_t)NLAY * FULLR * KK;
    const size_t nFULL  = (size_t)NLAY * FULLR * DD;

    us* wqkvr_bf[2]; us* wo_bf[2]; us* w1_bf[2]; us* w2_bf[2]; us* pk_bf[2]; us* full[2];
    for (int d = 0; d < 2; ++d) {
        wqkvr_bf[d] = (us*)alloc(nWqkvr * 2);
        wo_bf[d]    = (us*)alloc(nWo * 2);
        w1_bf[d]    = (us*)alloc(nW1 * 2);
        w2_bf[d]    = (us*)alloc(nW2 * 2);
        pk_bf[d]    = (us*)alloc(nPK * 2);
        full[d]     = (us*)alloc(nFULL * 2);
    }
    us*    Rbf     = (us*)alloc((size_t)FULLR * KK * 2);
    float* state_f = (float*)alloc((size_t)SEGL * DD * 4);
    float* q_f     = (float*)alloc((size_t)SEGL * KK * 4);
    us*    qu_bf   = (us*)alloc((size_t)SEGL * KK * 2);
    us*    qv_bf   = (us*)alloc((size_t)SEGL * KK * 2);
    us*    k_bf    = (us*)alloc((size_t)FULLR * KK * 2);
    us*    v_bf    = (us*)alloc((size_t)FULLR * KK * 2);
    float* S_f     = (float*)alloc((size_t)SEGL * FULLR * 4);
    us*    p_bf    = (us*)alloc((size_t)SEGL * FULLR * 2);
    us*    o_bf    = (us*)alloc((size_t)SEGL * KK * 2);
    float* ao_f    = (float*)alloc((size_t)SEGL * DD * 4);
    float* h_f     = (float*)alloc((size_t)SEGL * DD * 4);
    us*    h_bf    = (us*)alloc((size_t)SEGL * DD * 2);
    us*    f1_bf   = (us*)alloc((size_t)SEGL * FFD * 2);
    float* f2_f    = (float*)alloc((size_t)SEGL * DD * 4);
    us*    dummy_bf= (us*)alloc((size_t)SEGL * DD * 2);

    // --- weight conversion fp32 -> bf16 (each call; deterministic) ---
    for (int d = 0; d < 2; ++d) {
        f2bf_kernel<<<(unsigned)((nWqkvr + 255) / 256), 256, 0, stream>>>(Wqkvr[d], wqkvr_bf[d], (int)nWqkvr);
        f2bf_kernel<<<(unsigned)((nWo    + 255) / 256), 256, 0, stream>>>(Wo[d],    wo_bf[d],    (int)nWo);
        f2bf_kernel<<<(unsigned)((nW1    + 255) / 256), 256, 0, stream>>>(W1[d],    w1_bf[d],    (int)nW1);
        f2bf_kernel<<<(unsigned)((nW2    + 255) / 256), 256, 0, stream>>>(W2[d],    w2_bf[d],    (int)nW2);
    }

    // --- relative position matrix + pk = R @ Wr per layer/dir ---
    compute_R<<<FULLR, 256, 0, stream>>>(Rbf);
    for (int d = 0; d < 2; ++d)
        for (int j = 0; j < NLAY; ++j)
            gemm_wmma<0, 0, 0, 0, 1><<<dim3(KK / 64, FULLR / 64), 128, 0, stream>>>(
                Rbf, wqkvr_bf[d] + ((size_t)(j * 4 + 3)) * DD * KK,
                nullptr, pk_bf[d] + (size_t)j * FULLR * KK, FULLR, KK, DD);

    // --- zero memory ring (harness poisons ws; must be clean each call) ---
    for (int d = 0; d < 2; ++d)
        (void)hipMemsetAsync(full[d], 0, nFULL * 2, stream);

    const size_t rowBytes = (size_t)SEGL * DD * 2; // 512 rows * 512 bf16

    for (int d = 0; d < 2; ++d) {
        for (int seg = 0; seg < NSEGS; ++seg) {
            // shift memory ring: rows [512..1536) -> [0..1024), per layer
            if (seg > 0) {
                for (int j = 0; j < NLAY; ++j) {
                    us* fj = full[d] + (size_t)j * FULLR * DD;
                    (void)hipMemcpyAsync(fj,                   (char*)fj + rowBytes,     rowBytes, hipMemcpyDeviceToDevice, stream);
                    (void)hipMemcpyAsync((char*)fj + rowBytes, (char*)fj + 2 * rowBytes, rowBytes, hipMemcpyDeviceToDevice, stream);
                }
            }
            // embedding -> state (fp32) and layer-0 full tail (bf16)
            embed_gather<<<SEGL, 256, 0, stream>>>(src, emb, state_f,
                full[d] + (size_t)MEMR * DD, seg, d);

            const int n_valid = (seg < 2 ? seg : 2) * SEGL;

            for (int j = 0; j < NLAY; ++j) {
                us* fj       = full[d] + (size_t)j * FULLR * DD;
                us* state_bf = fj + (size_t)MEMR * DD;
                const us* wq = wqkvr_bf[d] + ((size_t)(j * 4 + 0)) * DD * KK;
                const us* wk = wqkvr_bf[d] + ((size_t)(j * 4 + 1)) * DD * KK;
                const us* wv = wqkvr_bf[d] + ((size_t)(j * 4 + 2)) * DD * KK;
                const us* pj = pk_bf[d] + (size_t)j * FULLR * KK;

                // q = state @ Wq
                gemm_wmma<0, 0, 0, 1, 0><<<dim3(8, 8), 128, 0, stream>>>(state_bf, wq, q_f, nullptr, SEGL, KK, DD);
                add_row_bf<<<1024, 256, 0, stream>>>(q_f, UV[d],      qu_bf);
                add_row_bf<<<1024, 256, 0, stream>>>(q_f, UV[d] + KK, qv_bf);
                // k, v = full @ Wk, full @ Wv
                gemm_wmma<0, 0, 0, 0, 1><<<dim3(8, 24), 128, 0, stream>>>(fj, wk, nullptr, k_bf, FULLR, KK, DD);
                gemm_wmma<0, 0, 0, 0, 1><<<dim3(8, 24), 128, 0, stream>>>(fj, wv, nullptr, v_bf, FULLR, KK, DD);
                // scores = (q+u)@k^T + (q+v)@pk^T
                gemm_wmma<1, 0, 0, 1, 0><<<dim3(24, 8), 128, 0, stream>>>(qu_bf, k_bf, S_f, nullptr, SEGL, FULLR, KK);
                gemm_wmma<1, 0, 1, 1, 0><<<dim3(24, 8), 128, 0, stream>>>(qv_bf, pj,  S_f, nullptr, SEGL, FULLR, KK);
                softmax_rows<<<SEGL, 256, 0, stream>>>(S_f, p_bf, n_valid);
                // out = attn @ val @ Wo
                gemm_wmma<0, 0, 0, 0, 1><<<dim3(8, 8), 128, 0, stream>>>(p_bf, v_bf, nullptr, o_bf, SEGL, DD, FULLR);
                gemm_wmma<0, 0, 0, 1, 0><<<dim3(8, 8), 128, 0, stream>>>(o_bf, wo_bf[d] + (size_t)j * KK * DD, ao_f, nullptr, SEGL, DD, KK);
                // h = LN(state + out)
                ln_residual<<<SEGL, 256, 0, stream>>>(state_f, ao_f,
                    LN[d] + (size_t)(j * 4 + 0) * DD, LN[d] + (size_t)(j * 4 + 1) * DD, h_f, h_bf);
                // ff = relu(h @ W1) @ W2
                gemm_wmma<0, 1, 0, 0, 1><<<dim3(32, 8), 128, 0, stream>>>(h_bf, w1_bf[d] + (size_t)j * DD * FFD, nullptr, f1_bf, SEGL, FFD, DD);
                gemm_wmma<0, 0, 0, 1, 0><<<dim3(8, 8), 128, 0, stream>>>(f1_bf, w2_bf[d] + (size_t)j * FFD * DD, f2_f, nullptr, SEGL, DD, FFD);
                // state = LN(h + ff) -> fp32 + bf16 into next layer's full tail
                us* next_bf = (j < NLAY - 1)
                    ? full[d] + (size_t)(j + 1) * FULLR * DD + (size_t)MEMR * DD
                    : dummy_bf;
                ln_residual<<<SEGL, 256, 0, stream>>>(h_f, f2_f,
                    LN[d] + (size_t)(j * 4 + 2) * DD, LN[d] + (size_t)(j * 4 + 3) * DD, state_f, next_bf);
            }
            write_out<<<SEGL, 256, 0, stream>>>(state_f, out, seg, d);
        }
    }
}
